// NeighborAttention_61572651155889
// MI455X (gfx1250) — matrix operand order repaired and checked by reading, not compiled, not run on hardware
//
#include <hip/hip_runtime.h>
#include <hip/hip_bf16.h>
#include <stdint.h>

typedef _Float16 half_t;
typedef __attribute__((ext_vector_type(16))) _Float16 v16h;
typedef __attribute__((ext_vector_type(8)))  _Float16 v8h;
typedef __attribute__((ext_vector_type(4)))  _Float16 v4h;
typedef __attribute__((ext_vector_type(8)))  float    v8f;

#define B_SZ 4096
#define N_NB 64
#define DXC  512
#define DDC  64
#define H_HD 8
#define DAC  64
#define DKV  640   // DX + 2*DD
#define HDA  512   // H*DA

// ---- WMMA fragment helpers (layouts per CDNA5 ISA 7.12.2) ----
// A (16x32 f16): lane<16 holds K {ko..ko+7, 16+ko..}, ko = (lane>>4)*8
__device__ __forceinline__ v16h make_afrag(const half_t* prow, int ko) {
  v8h lo = *(const v8h*)(prow + ko);
  v8h hi = *(const v8h*)(prow + 16 + ko);
  v16h r;
#pragma unroll
  for (int i = 0; i < 8; ++i) { r[i] = lo[i]; r[i + 8] = hi[i]; }
  return r;
}
// B (32x16 f16), K-major storage: lane%16 = n column, K {ko..ko+7, 16+ko..} strided by ld
__device__ __forceinline__ v16h make_bfrag(const half_t* p, int ld, int ko) {
  v16h r;
#pragma unroll
  for (int i = 0; i < 8; ++i) { r[i] = p[(long)(ko + i) * ld]; r[i + 8] = p[(long)(16 + ko + i) * ld]; }
  return r;
}

// ---- K0: weight transposes/f16 conversion + renormed embedding tables ----
__global__ void __launch_bounds__(256)
prep_kernel(const float* __restrict__ xi, const float* __restrict__ Wq,
            const float* __restrict__ Wk, const float* __restrict__ Wv,
            const float* __restrict__ Wo, const float* __restrict__ degk,
            const float* __restrict__ degv, const float* __restrict__ distk,
            const float* __restrict__ distv,
            half_t* Xi16, half_t* WqT, half_t* Wk16, half_t* WvT, half_t* WoT,
            half_t* Tdk, half_t* Tdv, half_t* Ttk, half_t* Ttv)
{
  const long NXI = (long)B_SZ * DXC;
  const long NWQ = (long)HDA * DXC;
  const long NWK = (long)HDA * DKV;
  const long NWV = NWK;
  const long total = NXI + NWQ + NWK + NWV + NWQ;
  for (long i = (long)blockIdx.x * 256 + threadIdx.x; i < total; i += (long)gridDim.x * 256) {
    long t = i;
    if (t < NXI) { Xi16[t] = (half_t)xi[t]; continue; }
    t -= NXI;
    if (t < NWQ) { long c = t / DXC, f = t % DXC; WqT[t] = (half_t)Wq[f * DXC + c]; continue; }
    t -= NWQ;
    if (t < NWK) { Wk16[t] = (half_t)Wk[t]; continue; }
    t -= NWK;
    if (t < NWV) { long k = t / HDA, f = t % HDA; WvT[t] = (half_t)Wv[f * DKV + k]; continue; }
    t -= NWV;
    { long f = t / DXC, c = t % DXC; WoT[t] = (half_t)Wo[c * HDA + f]; }
  }
  // embedding tables: fold max_norm=1.0 (inf-norm) renorm into f16 tables
  const int NROWS = 1001 + 1001 + 9 + 9;
  for (int r = blockIdx.x * 256 + threadIdx.x; r < NROWS; r += gridDim.x * 256) {
    const float* src; half_t* dst; int rr = r;
    if (rr < 1001)      {            src = degk  + rr * DDC; dst = Tdk + rr * DDC; }
    else if (rr < 2002) { rr -= 1001; src = degv  + rr * DDC; dst = Tdv + rr * DDC; }
    else if (rr < 2011) { rr -= 2002; src = distk + rr * DDC; dst = Ttk + rr * DDC; }
    else                { rr -= 2011; src = distv + rr * DDC; dst = Ttv + rr * DDC; }
    float mx = 0.f;
    for (int d = 0; d < DDC; ++d) mx = fmaxf(mx, fabsf(src[d]));
    float sc = fminf(1.f, 1.f / fmaxf(mx, 1e-12f));
    for (int d = 0; d < DDC; ++d) dst[d] = (half_t)(src[d] * sc);
  }
}

// ---- Generic batched WMMA GEMM: C = scale * A(f16,row-major) @ B(f16,K-major) [+res] ----
// Block tile 128(M) x 128(N); 8 waves; wave -> 32x64 (2 A-frags share each B-frag)
__global__ void __launch_bounds__(256)
wmma_gemm_f16(const half_t* __restrict__ A, int lda, long aOffZ,
              const half_t* __restrict__ Bm, int ldb, long bOffZ,
              half_t* __restrict__ C16, float* __restrict__ C32,
              int ldc, long cOffZ, const float* __restrict__ res,
              int M, int N, int K, float scale, int outF16)
{
  const int lane = threadIdx.x & 31;
  const int wave = threadIdx.x >> 5;
  A  += (long)blockIdx.z * aOffZ;
  Bm += (long)blockIdx.z * bOffZ;
  const long coff = (long)blockIdx.z * cOffZ;

  const int m0 = blockIdx.x * 128 + (wave & 3) * 32;
  const int n0 = blockIdx.y * 128 + (wave >> 2) * 64;
  if (m0 >= M || n0 >= N) return;   // no barriers in this kernel

  const int row = m0 + (lane & 15);
  const int ko  = (lane >> 4) << 3;

  v8f zerov = {};
  v8f acc0[4], acc1[4];
#pragma unroll
  for (int j = 0; j < 4; ++j) { acc0[j] = zerov; acc1[j] = zerov; }

  for (int kc = 0; kc < K; kc += 32) {
    v16h a0 = make_afrag(A + (long)row * lda + kc, ko);
    v16h a1 = make_afrag(A + (long)(row + 16) * lda + kc, ko);
    const half_t* bp = Bm + (long)kc * ldb + n0 + (lane & 15);
#pragma unroll
    for (int j = 0; j < 4; ++j) {
      v16h b = make_bfrag(bp + 16 * j, ldb, ko);
      acc0[j] = __builtin_amdgcn_wmma_f32_16x16x32_f16(false, a0, false, b,
                                                       (short)0, acc0[j], false, false);
      acc1[j] = __builtin_amdgcn_wmma_f32_16x16x32_f16(false, a1, false, b,
                                                       (short)0, acc1[j], false, false);
    }
  }
  const int madd = (lane >> 4) << 3;
  const int ncol = lane & 15;
#pragma unroll
  for (int j = 0; j < 4; ++j) {
    int n = n0 + 16 * j + ncol;
#pragma unroll
    for (int r = 0; r < 8; ++r) {
      {
        int m = m0 + r + madd;
        float v = acc0[j][r] * scale;
        long idx = coff + (long)m * ldc + n;
        if (outF16) C16[idx] = (half_t)v;
        else { if (res) v += res[(long)m * ldc + n]; C32[idx] = v; }
      }
      {
        int m = m0 + 16 + r + madd;
        float v = acc1[j][r] * scale;
        long idx = coff + (long)m * ldc + n;
        if (outF16) C16[idx] = (half_t)v;
        else { if (res) v += res[(long)m * ldc + n]; C32[idx] = v; }
      }
    }
  }
}

// ---- K3: fused per-element attention core ----
// LDS: Xs[64][648] = [xj | E_dv | E_tv] f16 ; EKs[64][136] = [E_dk | E_tk]
//      qkTs[640][24] = QK^T (K-major B) ; Als[16][72] = alpha ; sTs[64][20] = scores^T
__global__ void __launch_bounds__(256)
attn_core(const float* __restrict__ xj, const int* __restrict__ deg,
          const int* __restrict__ dist, const uint8_t* __restrict__ mask,
          const half_t* __restrict__ Tdk, const half_t* __restrict__ Tdv,
          const half_t* __restrict__ Ttk, const half_t* __restrict__ Ttv,
          half_t* __restrict__ QKZ,       // in: QK (b,8,640) f16; out: Z in-place
          float* __restrict__ outAlpha)
{
  extern __shared__ char smem_raw[];
  half_t* Xs   = (half_t*)smem_raw;            // 64*648
  half_t* EKs  = Xs   + 64 * 648;              // 64*136
  half_t* qkTs = EKs  + 64 * 136;              // 640*24
  half_t* Als  = qkTs + 640 * 24;              // 16*72
  float*  sTs  = (float*)(Als + 16 * 72);      // 64*20

  const int b = blockIdx.x;
  const int tid = threadIdx.x;
  const int lane = tid & 31;
  const int wave = tid >> 5;

  // P0: stage QK^T (B matrix for scores, K-major over h), 8 halfs per step
  const half_t* qkg = QKZ + (long)b * (H_HD * DKV);
  for (int i = tid * 8; i < H_HD * DKV; i += 256 * 8) {
    int h = i / DKV, k = i % DKV;      // k multiple of 8
    v8h v = *(const v8h*)(qkg + i);
#pragma unroll
    for (int j = 0; j < 8; ++j) qkTs[(k + j) * 24 + h] = v[j];
  }
  for (int i = tid; i < 16 * 72; i += 256) Als[i] = (half_t)0.f;

  // P1: xj -> f16 tile, vectorized b128 loads (the HBM-critical stream)
  const float* xjb = xj + (long)b * (N_NB * DXC);
  for (int i = tid * 4; i < N_NB * DXC; i += 256 * 4) {
    if (i + 8192 < N_NB * DXC) __builtin_prefetch(xjb + i + 8192, 0, 0);
    float4 f = *(const float4*)(xjb + i);
    v4h h4; h4[0] = (half_t)f.x; h4[1] = (half_t)f.y; h4[2] = (half_t)f.z; h4[3] = (half_t)f.w;
    int n = i >> 9, c = i & 511;       // c multiple of 4 -> 8B-aligned LDS store
    *(v4h*)(Xs + n * 648 + c) = h4;
  }
  // P2: gather renormed embeddings, 8 halfs (b128) per step
  const int* degb  = deg  + b * N_NB;
  const int* distb = dist + b * N_NB;
  for (int i = tid * 8; i < N_NB * DDC; i += 256 * 8) {
    int n = i >> 6, d = i & 63;        // d multiple of 8
    int dg = degb[n];  dg = dg > 1000 ? 1000 : dg;
    int dt = distb[n]; dt = dt > 8 ? 8 : dt;
    *(v8h*)(Xs + n * 648 + 512 + d) = *(const v8h*)(Tdv + dg * DDC + d);
    *(v8h*)(Xs + n * 648 + 576 + d) = *(const v8h*)(Ttv + dt * DDC + d);
    *(v8h*)(EKs + n * 136 + d)      = *(const v8h*)(Tdk + dg * DDC + d);
    *(v8h*)(EKs + n * 136 + 64 + d) = *(const v8h*)(Ttk + dt * DDC + d);
  }
  __syncthreads();

  // P3: scores^T (64x16) = [X | EK](64x640) @ qkT(640x16); waves 0..3, tile each
  if (wave < 4) {
    const int m0  = wave * 16;
    const int row = m0 + (lane & 15);
    const int ko3 = (lane >> 4) << 3;
    v8f acc = {};
    for (int kc = 0; kc < DKV; kc += 32) {
      const half_t* ap = (kc < 512) ? (Xs + row * 648 + kc)
                                    : (EKs + row * 136 + (kc - 512));
      v16h a = make_afrag(ap, ko3);
      v16h bfr = make_bfrag(qkTs + kc * 24 + (lane & 15), 24, ko3);
      acc = __builtin_amdgcn_wmma_f32_16x16x32_f16(false, a, false, bfr,
                                                   (short)0, acc, false, false);
    }
    const int madd = (lane >> 4) << 3;
    const int hh = lane & 15;
#pragma unroll
    for (int r = 0; r < 8; ++r)
      sTs[(m0 + r + madd) * 20 + hh] = acc[r];
  }
  __syncthreads();

  // P4: masked softmax, one wave32 per head (mask[:,0] forced true)
  {
    const int h = wave;
    const uint8_t* mb = mask + b * N_NB;
    int n1 = lane, n2 = lane + 32;
    float s1 = sTs[n1 * 20 + h];
    float s2 = sTs[n2 * 20 + h];
    s1 = ((n1 == 0) || mb[n1]) ? s1 : -10000000.0f;
    s2 = (mb[n2])              ? s2 : -10000000.0f;
    float mx = fmaxf(s1, s2);
#pragma unroll
    for (int off = 16; off > 0; off >>= 1) mx = fmaxf(mx, __shfl_xor(mx, off, 32));
    float e1 = __expf(s1 - mx), e2 = __expf(s2 - mx);
    float sm = e1 + e2;
#pragma unroll
    for (int off = 16; off > 0; off >>= 1) sm += __shfl_xor(sm, off, 32);
    float inv = 1.0f / sm;
    float a1 = e1 * inv, a2 = e2 * inv;
    float* oa = outAlpha + (long)b * (H_HD * N_NB) + h * N_NB;
    oa[n1] = a1; oa[n2] = a2;
    Als[h * 72 + n1] = (half_t)a1;
    Als[h * 72 + n2] = (half_t)a2;
  }
  __syncthreads();

  // P5: Z(16x640) = Alpha(16x64) @ [X|E_dv|E_tv](64x640); 40 N-tiles over 8 waves
  half_t* zg = QKZ + (long)b * (H_HD * DKV);
  const int arow = lane & 15;
  const int ko5  = (lane >> 4) << 3;
  for (int t = wave; t < 40; t += 8) {
    const int n0 = t * 16;
    v8f acc = {};
#pragma unroll
    for (int kc = 0; kc < 64; kc += 32) {
      v16h a   = make_afrag(Als + arow * 72 + kc, ko5);
      v16h bfr = make_bfrag(Xs + kc * 648 + n0 + (lane & 15), 648, ko5);
      acc = __builtin_amdgcn_wmma_f32_16x16x32_f16(false, a, false, bfr,
                                                   (short)0, acc, false, false);
    }
    if (lane < 16) {   // rows m=0..7 (real heads) live in lanes 0-15
#pragma unroll
      for (int r = 0; r < 8; ++r)
        zg[r * DKV + n0 + lane] = (half_t)acc[r];
    }
  }
}

// ---- K6: LayerNorm (residual already in Ylin) ----
__global__ void __launch_bounds__(256)
ln_kernel(const float* __restrict__ Ylin, const float* __restrict__ gamma,
          const float* __restrict__ beta, float* __restrict__ Yout)
{
  __shared__ float red[8];
  const int b = blockIdx.x, t = threadIdx.x;
  const int lane = t & 31, wave = t >> 5;
  const float* yr = Ylin + (long)b * DXC;
  float v0 = yr[t], v1 = yr[t + 256];
  float s = v0 + v1;
#pragma unroll
  for (int off = 16; off > 0; off >>= 1) s += __shfl_xor(s, off, 32);
  if (lane == 0) red[wave] = s;
  __syncthreads();
  float tot = 0.f;
#pragma unroll
  for (int i = 0; i < 8; ++i) tot += red[i];
  float mu = tot * (1.0f / DXC);
  float d0 = v0 - mu, d1 = v1 - mu;
  float q = d0 * d0 + d1 * d1;
#pragma unroll
  for (int off = 16; off > 0; off >>= 1) q += __shfl_xor(q, off, 32);
  __syncthreads();
  if (lane == 0) red[wave] = q;
  __syncthreads();
  float vq = 0.f;
#pragma unroll
  for (int i = 0; i < 8; ++i) vq += red[i];
  float rstd = rsqrtf(vq * (1.0f / DXC) + 1e-5f);
  float* yo = Yout + (long)b * DXC;
  yo[t]       = d0 * rstd * gamma[t]       + beta[t];
  yo[t + 256] = d1 * rstd * gamma[t + 256] + beta[t + 256];
}

extern "C" void kernel_launch(void* const* d_in, const int* in_sizes, int n_in,
                              void* d_out, int out_size, void* d_ws, size_t ws_size,
                              hipStream_t stream)
{
  (void)in_sizes; (void)n_in; (void)out_size; (void)ws_size;
  const float*   xi    = (const float*)d_in[0];
  const float*   xj    = (const float*)d_in[1];
  const int*     deg   = (const int*)d_in[2];
  const int*     dist  = (const int*)d_in[3];
  const uint8_t* mask  = (const uint8_t*)d_in[4];
  const float*   Wq    = (const float*)d_in[5];
  const float*   Wk    = (const float*)d_in[6];
  const float*   Wv    = (const float*)d_in[7];
  const float*   Wo    = (const float*)d_in[8];
  const float*   degk  = (const float*)d_in[9];
  const float*   degv  = (const float*)d_in[10];
  const float*   distk = (const float*)d_in[11];
  const float*   distv = (const float*)d_in[12];
  const float*   lng   = (const float*)d_in[13];
  const float*   lnb   = (const float*)d_in[14];

  char* ws = (char*)d_ws;
  size_t off = 0;
  auto take = [&](size_t bytes) { char* p = ws + off; off += (bytes + 255) & ~(size_t)255; return p; };
  half_t* WqT  = (half_t*)take((size_t)512 * 512 * 2);
  half_t* Wk16 = (half_t*)take((size_t)512 * 640 * 2);
  half_t* WvT  = (half_t*)take((size_t)640 * 512 * 2);
  half_t* WoT  = (half_t*)take((size_t)512 * 512 * 2);
  half_t* Tdk  = (half_t*)take((size_t)1001 * 64 * 2);
  half_t* Tdv  = (half_t*)take((size_t)1001 * 64 * 2);
  half_t* Ttk  = (half_t*)take((size_t)9 * 64 * 2);
  half_t* Ttv  = (half_t*)take((size_t)9 * 64 * 2);
  half_t* Xi16 = (half_t*)take((size_t)4096 * 512 * 2);
  half_t* Qbuf = (half_t*)take((size_t)4096 * 512 * 2);      // Q, later reused for attn
  half_t* QKZ  = (half_t*)take((size_t)4096 * 8 * 640 * 2);  // QK, reused in-place for Z
  float*  Ylin = (float*)take((size_t)4096 * 512 * 4);

  float* yOut = (float*)d_out;
  float* aOut = yOut + (size_t)4096 * 512;

  prep_kernel<<<1024, 256, 0, stream>>>(xi, Wq, Wk, Wv, Wo, degk, degv, distk, distv,
                                        Xi16, WqT, Wk16, WvT, WoT, Tdk, Tdv, Ttk, Ttv);

  // K1: Q = xi @ Wq^T  (M=4096,N=512,K=512)
  dim3 g1(4096 / 128, 512 / 128, 1);
  wmma_gemm_f16<<<g1, 256, 0, stream>>>(Xi16, 512, 0, WqT, 512, 0,
                                        Qbuf, nullptr, 512, 0, nullptr,
                                        4096, 512, 512, 1.0f, 1);

  // K2: QK_h = (Q_h / TEMP) @ Wk_h  per head (M=4096,N=640,K=64)
  dim3 g2(4096 / 128, 640 / 128, 8);
  wmma_gemm_f16<<<g2, 256, 0, stream>>>(Qbuf, 512, 64, Wk16, 640, 64L * 640,
                                        QKZ, nullptr, 5120, 640, nullptr,
                                        4096, 640, 64, 0.125f, 1);

  // K3: fused scores + softmax + alpha-weighted feature sum (Z), one WG per element
  size_t smem = (size_t)(64 * 648 + 64 * 136 + 640 * 24 + 16 * 72) * 2 + (size_t)(64 * 20) * 4;
  attn_core<<<4096, 256, smem, stream>>>(xj, deg, dist, mask, Tdk, Tdv, Ttk, Ttv, QKZ, aOut);

  // K4: attn_h = Z_h @ WvT[:,64h:64h+64]  per head (M=4096,N=64,K=640)
  dim3 g4(4096 / 128, 1, 8);
  wmma_gemm_f16<<<g4, 256, 0, stream>>>(QKZ, 5120, 640, WvT, 512, 64,
                                        Qbuf, nullptr, 512, 64, nullptr,
                                        4096, 64, 640, 1.0f, 1);

  // K5: Ylin = xi + attn @ Wo^T  (M=4096,N=512,K=512), f32 out with residual
  dim3 g5(4096 / 128, 512 / 128, 1);
  wmma_gemm_f16<<<g5, 256, 0, stream>>>(Qbuf, 512, 0, WoT, 512, 0,
                                        nullptr, Ylin, 512, 0, xi,
                                        4096, 512, 512, 1.0f, 0);

  // K6: LayerNorm -> y
  ln_kernel<<<4096, 256, 0, stream>>>(Ylin, lng, lnb, yOut);
}
